// SelfAttention_90744069030056
// MI455X (gfx1250) — matrix operand order repaired and checked by reading, not compiled, not run on hardware
//
#include <hip/hip_runtime.h>
#include <hip/hip_bf16.h>

// ---------------------------------------------------------------------------
// MI455X (gfx1250).
// Algebraic collapse: s2 = (q k^T) a = code * (W_q^T W_k) * (code^T text) * W_a^T
// -> ~86 GFLOP instead of ~350, no [B,S,S] intermediate.
// All GEMMs: v_wmma_f32_16x16x32_bf16, double-buffered LDS, async
// global->LDS (ASYNCcnt) for bf16 operands, register-prefetch+convert for f32.
// Producers store transposed (vT, Gt, NT) so all bf16 consumer operands are
// contiguous global->LDS copies (async-eligible, no transpose scatter).
// ---------------------------------------------------------------------------

typedef __bf16 bf16_t;
typedef __attribute__((ext_vector_type(16))) __bf16 v16bf;
typedef __attribute__((ext_vector_type(8)))  float  v8f;
typedef __attribute__((ext_vector_type(4)))  unsigned int v4u;   // 16-byte chunk
typedef __attribute__((ext_vector_type(4)))  int v4i;

#define LPITCH 40   // LDS row pitch in bf16 elems: 80B, 16B-aligned, bank-spread

#if __has_builtin(__builtin_amdgcn_global_load_async_to_lds_b128)
#define HAVE_ASYNC 1
#else
#define HAVE_ASYNC 0
#endif

static __device__ inline void wait_async0() {
#if __has_builtin(__builtin_amdgcn_s_wait_asynccnt)
    __builtin_amdgcn_s_wait_asynccnt(0);
#else
    asm volatile("s_wait_asynccnt 0x0" ::: "memory");
#endif
}

#if HAVE_ASYNC
// Copy 32 contiguous bytes global -> LDS via the async path (ASYNCcnt).
static __device__ inline void async_copy32B(const void* g, void* l) {
    auto gp = (__attribute__((address_space(1))) v4i*)(void*)g;
    auto lp = (__attribute__((address_space(3))) v4i*)l;
    __builtin_amdgcn_global_load_async_to_lds_b128(gp, lp, 0, 0);
    __builtin_amdgcn_global_load_async_to_lds_b128(gp + 1, lp + 1, 0, 0);
}
#endif

static __device__ inline __bf16 to_bf(float x)  { return (__bf16)x; }
static __device__ inline __bf16 to_bf(__bf16 x) { return x; }

union FragAB { v16bf v; v4u u[2]; };

// ---------------------------------------------------------------------------
// Tiled GEMM:  C (+)= alpha * op(A)[M,K] * op(B)[K,N]
//   TA: A[m][k] = storedA[k*lda+m]  else storedA[m*lda+k]
//   TB: B[k][n] = storedB[n*ldb+k]  else storedB[k*ldb+n]
//   OTRANS: store C transposed: C[n*ldc+m]
// Block 256 threads (8 waves), tile BM=BN=128, BK=32, double-buffered LDS.
// Wave grid 4(m)x2(n): each wave 2x4 wmma tiles.  Dims multiples of 128/32.
// bf16 A(!TA) and bf16 B(TB) operands go through global_load_async_to_lds.
// ---------------------------------------------------------------------------
template<bool TA, bool TB, typename AT, typename BT, bool OBF16, bool OTRANS>
__global__ __launch_bounds__(256)
void gemm_wmma(const AT* __restrict__ A, const BT* __restrict__ B,
               void* __restrict__ Cv,
               int M, int N, int K,
               long lda, long ldb, long ldc,
               long sA, long sB, long sC, float alpha)
{
    (void)M; (void)N;
    constexpr bool A_ASYNC = (HAVE_ASYNC != 0) && (!TA) && __is_same(AT, bf16_t);
    constexpr bool B_ASYNC = (HAVE_ASYNC != 0) && ( TB) && __is_same(BT, bf16_t);
    constexpr bool ANY_ASYNC = A_ASYNC || B_ASYNC;

    __shared__ __align__(16) __bf16 lA[2][128 * LPITCH];   // [m][k]
    __shared__ __align__(16) __bf16 lB[2][128 * LPITCH];   // [n][k]

    const AT* Ab = A + (size_t)blockIdx.z * sA;
    const BT* Bb = B + (size_t)blockIdx.z * sB;
    const int m0 = blockIdx.y * 128;
    const int n0 = blockIdx.x * 128;

    const int t    = threadIdx.x;
    const int lane = t & 31;
    const int wid  = t >> 5;
    const int wm   = wid >> 1;          // 0..3
    const int wn   = wid & 1;           // 0..1
    const int ml   = lane & 15;
    const int h    = lane >> 4;

    AT aReg[16];
    BT bReg[16];

    v8f acc[2][4];
    #pragma unroll
    for (int i = 0; i < 2; ++i)
        #pragma unroll
        for (int j = 0; j < 4; ++j)
            acc[i][j] = (v8f)0.0f;

    // ---- stage issue: async writes LDS[st] directly; reg path -> aReg/bReg
    auto issueA = [&](int k0, int st) {
        if constexpr (A_ASYNC) {
            int row = t >> 1, cb = (t & 1) * 16;
            async_copy32B(Ab + (size_t)(m0 + row) * lda + (k0 + cb),
                          &lA[st][row * LPITCH + cb]);
        } else if constexpr (!TA) {
            int row = t >> 1, cb = (t & 1) * 16;
            const AT* src = Ab + (size_t)(m0 + row) * lda + (k0 + cb);
            #pragma unroll
            for (int i = 0; i < 16; ++i) aReg[i] = src[i];
        } else {
            int kk = t >> 3, mb = (t & 7) * 16;
            const AT* src = Ab + (size_t)(k0 + kk) * lda + (m0 + mb);
            #pragma unroll
            for (int i = 0; i < 16; ++i) aReg[i] = src[i];
        }
    };
    auto issueB = [&](int k0, int st) {
        if constexpr (B_ASYNC) {
            int nn = t >> 1, cb = (t & 1) * 16;
            async_copy32B(Bb + (size_t)(n0 + nn) * ldb + (k0 + cb),
                          &lB[st][nn * LPITCH + cb]);
        } else if constexpr (TB) {
            int nn = t >> 1, cb = (t & 1) * 16;
            const BT* src = Bb + (size_t)(n0 + nn) * ldb + (k0 + cb);
            #pragma unroll
            for (int i = 0; i < 16; ++i) bReg[i] = src[i];
        } else {
            int kk = t >> 3, nb = (t & 7) * 16;
            const BT* src = Bb + (size_t)(k0 + kk) * ldb + (n0 + nb);
            #pragma unroll
            for (int i = 0; i < 16; ++i) bReg[i] = src[i];
        }
    };
    auto storeA = [&](int st) {
        if constexpr (!A_ASYNC) {
            if constexpr (!TA) {
                int row = t >> 1, cb = (t & 1) * 16;
                #pragma unroll
                for (int i = 0; i < 16; ++i) lA[st][row * LPITCH + cb + i] = to_bf(aReg[i]);
            } else {
                int kk = t >> 3, mb = (t & 7) * 16;
                #pragma unroll
                for (int i = 0; i < 16; ++i) lA[st][(mb + i) * LPITCH + kk] = to_bf(aReg[i]);
            }
        }
    };
    auto storeB = [&](int st) {
        if constexpr (!B_ASYNC) {
            if constexpr (TB) {
                int nn = t >> 1, cb = (t & 1) * 16;
                #pragma unroll
                for (int i = 0; i < 16; ++i) lB[st][nn * LPITCH + cb + i] = to_bf(bReg[i]);
            } else {
                int kk = t >> 3, nb = (t & 7) * 16;
                #pragma unroll
                for (int i = 0; i < 16; ++i) lB[st][(nb + i) * LPITCH + kk] = to_bf(bReg[i]);
            }
        }
    };
    // ---- 8 wmma from LDS stage st (ISA lane layouts) ----------------------
    auto compute = [&](int st) {
        FragAB af[2];
        #pragma unroll
        for (int i = 0; i < 2; ++i) {
            int row = wm * 32 + i * 16 + ml;
            af[i].u[0] = *(const v4u*)&lA[st][row * LPITCH + h * 8];
            af[i].u[1] = *(const v4u*)&lA[st][row * LPITCH + 16 + h * 8];
        }
        FragAB bfg[4];
        #pragma unroll
        for (int j = 0; j < 4; ++j) {
            int nr = wn * 64 + j * 16 + ml;
            bfg[j].u[0] = *(const v4u*)&lB[st][nr * LPITCH + h * 8];
            bfg[j].u[1] = *(const v4u*)&lB[st][nr * LPITCH + 16 + h * 8];
        }
        #pragma unroll
        for (int i = 0; i < 2; ++i)
            #pragma unroll
            for (int j = 0; j < 4; ++j)
                acc[i][j] = __builtin_amdgcn_wmma_f32_16x16x32_bf16(
                    false, af[i].v, false, bfg[j].v,
                    (short)0, acc[i][j], false, false);
    };

    // ---- software pipeline: 1 barrier per 8 wmma --------------------------
    issueA(0, 0); issueB(0, 0);
    storeA(0);    storeB(0);
    if (ANY_ASYNC) wait_async0();
    __syncthreads();

    int buf = 0;
    for (int k0 = 32; k0 < K; k0 += 32) {
        issueA(k0, buf ^ 1); issueB(k0, buf ^ 1);   // next stage in flight
        compute(buf);                                // wmma on current stage
        storeA(buf ^ 1);     storeB(buf ^ 1);        // reg-path deposit
        if (ANY_ASYNC) wait_async0();
        __syncthreads();
        buf ^= 1;
    }
    compute(buf);

    // ---- epilogue: C VGPR r holds M = r + 8*h, N = ml ---------------------
    float*  Cf = (float*)Cv  + (size_t)blockIdx.z * sC;
    bf16_t* Cb = (bf16_t*)Cv + (size_t)blockIdx.z * sC;
    #pragma unroll
    for (int i = 0; i < 2; ++i)
        #pragma unroll
        for (int j = 0; j < 4; ++j) {
            int n = n0 + wn * 64 + j * 16 + ml;
            #pragma unroll
            for (int r = 0; r < 8; ++r) {
                int m = m0 + wm * 32 + i * 16 + h * 8 + r;
                float val = acc[i][j][r] * alpha;
                size_t idx = OTRANS ? (size_t)n * ldc + m : (size_t)m * ldc + n;
                if (OBF16) Cb[idx] = (__bf16)val;
                else       Cf[idx] = val;
            }
        }
}

// ---------------------------------------------------------------------------
// Row softmax: one 256-thread block per row of length D; writes bf16.
// ---------------------------------------------------------------------------
__global__ __launch_bounds__(256)
void softmax_rows(const float* __restrict__ s2, bf16_t* __restrict__ attn, int D)
{
    __shared__ float red[256];
    const size_t base = (size_t)blockIdx.x * D;
    const int t = threadIdx.x;

    float mx = -INFINITY;
    for (int i = t; i < D; i += 256) mx = fmaxf(mx, s2[base + i]);
    red[t] = mx; __syncthreads();
    for (int s = 128; s > 0; s >>= 1) {
        if (t < s) red[t] = fmaxf(red[t], red[t + s]);
        __syncthreads();
    }
    mx = red[0]; __syncthreads();

    float sum = 0.0f;
    for (int i = t; i < D; i += 256) sum += __expf(s2[base + i] - mx);
    red[t] = sum; __syncthreads();
    for (int s = 128; s > 0; s >>= 1) {
        if (t < s) red[t] += red[t + s];
        __syncthreads();
    }
    const float inv = 1.0f / red[0];

    for (int i = t; i < D; i += 256)
        attn[base + i] = (__bf16)(__expf(s2[base + i] - mx) * inv);
}

// ---------------------------------------------------------------------------
// Host launch
// ---------------------------------------------------------------------------
extern "C" void kernel_launch(void* const* d_in, const int* in_sizes, int n_in,
                              void* d_out, int out_size, void* d_ws, size_t ws_size,
                              hipStream_t stream)
{
    (void)in_sizes; (void)n_in; (void)out_size; (void)ws_size;

    constexpr int Bn = 16, S = 2048, E = 256, D = 2048;
    constexpr long SE = (long)S * E;      // 524288
    constexpr long EE = (long)E * E;      // 65536
    constexpr long ED = (long)E * D;      // 524288
    constexpr long SD = (long)S * D;      // 4194304

    const float* code = (const float*)d_in[0];
    const float* text = (const float*)d_in[1];
    const float* Wv   = (const float*)d_in[2];
    const float* Wk   = (const float*)d_in[3];
    const float* Wq   = (const float*)d_in[4];
    const float* Wa   = (const float*)d_in[5];
    float* out        = (float*)d_out;

    // workspace layout (bytes) -- transposed producers
    char* ws = (char*)d_ws;
    bf16_t* vT   = (bf16_t*)(ws);                         // [B,E,S] bf16   16.8 MB
    bf16_t* Gt   = (bf16_t*)(ws + (size_t)16777216);      // [B,E,E] bf16    2.1 MB
    bf16_t* Wqk  = (bf16_t*)(ws + (size_t)18874368);      // [E,E]   bf16    0.13 MB
    bf16_t* T    = (bf16_t*)(ws + (size_t)19005440);      // [B,E,E] bf16    2.1 MB
    bf16_t* NT   = (bf16_t*)(ws + (size_t)21102592);      // [B,D,E] bf16   16.8 MB
    bf16_t* attn = (bf16_t*)(ws + (size_t)37879808);      // [B,S,D] bf16  134.2 MB
    float*  s2   = (float*) (ws + (size_t)172097536);     // [B,S,D] f32   268.4 MB

    dim3 blk(256);
    auto grd = [](int M, int N, int bat) { return dim3(N / 128, M / 128, bat); };

    // 1) vT[b] = (code[b] @ W_v^T)^T                       [B,E,S]
    gemm_wmma<false, true, float, float, true, true><<<grd(S, E, Bn), blk, 0, stream>>>(
        code, Wv, vT, S, E, E, E, E, S, SE, 0, SE, 1.0f);

    // 2) Gt[b] = (code[b]^T @ text[b])^T                   [B,E,E]
    gemm_wmma<true, false, float, float, true, true><<<grd(E, E, Bn), blk, 0, stream>>>(
        code, text, Gt, E, E, S, E, E, E, SE, SE, EE, 1.0f);

    // 3) Wqk = W_q^T @ W_k                                 [E,E]
    gemm_wmma<true, false, float, float, true, false><<<grd(E, E, 1), blk, 0, stream>>>(
        Wq, Wk, Wqk, E, E, E, E, E, E, 0, 0, 0, 1.0f);

    // 4) T[b] = Wqk @ G[b]    (A async, B async via Gt)    [B,E,E]
    gemm_wmma<false, true, bf16_t, bf16_t, true, false><<<grd(E, E, Bn), blk, 0, stream>>>(
        Wqk, Gt, T, E, E, E, E, E, E, 0, EE, EE, 1.0f);

    // 5) NT[b] = (T[b] @ W_a^T)^T    (A async)             [B,D,E]
    gemm_wmma<false, true, bf16_t, float, true, true><<<grd(E, D, Bn), blk, 0, stream>>>(
        T, Wa, NT, E, D, E, E, E, E, EE, 0, ED, 1.0f);

    // 6) s2[b] = code[b] @ N[b] / sqrt(E)  (B async via NT) [B,S,D] f32
    gemm_wmma<false, true, float, bf16_t, false, false><<<grd(S, D, Bn), blk, 0, stream>>>(
        code, NT, s2, S, D, E, E, E, D, SE, ED, SD, 0.0625f);

    // 7) attn = softmax_rows(s2)                           [B,S,D] bf16
    softmax_rows<<<dim3(Bn * S), blk, 0, stream>>>(s2, attn, D);

    // 8) out[b] = attn[b] @ v[b]   (A async, B async via vT) [B,S,E] f32
    gemm_wmma<false, true, bf16_t, bf16_t, false, false><<<grd(S, E, Bn), blk, 0, stream>>>(
        attn, vT, out, S, E, D, D, S, E, SD, SE, SE, 1.0f);
}
